// MambaRegressor_68616397521124
// MI455X (gfx1250) — matrix operand order repaired and checked by reading, not compiled, not run on hardware
//
#include <hip/hip_runtime.h>
#include <hip/hip_bf16.h>

// ---------------- problem constants ----------------
#define DIN    32
#define KSEQ   512
#define PDIM   64
#define DMODEL 128
#define NLAYER 4
#define PLEN   8
#define PSTRIDE 4
#define NPATCH 127           // (512-8)/4+1
#define BATCH  32
#define DINNER 256           // DI = 2*DM
#define DSTATE 16
#define DCONV  4
#define DTRANK 8
#define MSEQ   (BATCH*PDIM)  // 2048 sequences
#define ROWS   (MSEQ*NPATCH) // 260096 flattened (m,t) rows
#define XPAD   48            // xproj output dim 40 padded to 48

typedef __attribute__((ext_vector_type(16))) __bf16 v16bf;
typedef __attribute__((ext_vector_type(8)))  float  v8f;
typedef unsigned int uv4 __attribute__((ext_vector_type(4)));

union ABFrag { uv4 u[2]; v16bf v; };

// ---------------- weight prep: transpose fp32 -> bf16 (optionally zero-padded) ----------------
// src: (Kd, Nsrc) row-major ; dst: (Npad, Kd) row-major, rows >= Nsrc are zero
__global__ void transpose_bf16_kernel(const float* __restrict__ src,
                                      __hip_bfloat16* __restrict__ dst,
                                      int Kd, int Nsrc, int Npad) {
  long gid = (long)blockIdx.x * blockDim.x + threadIdx.x;
  long total = (long)Npad * Kd;
  if (gid >= total) return;
  int n = (int)(gid / Kd);
  int k = (int)(gid % Kd);
  float v = (n < Nsrc) ? src[(long)k * Nsrc + n] : 0.0f;
  dst[gid] = __float2bfloat16(v);
}

// ---------------- frontend: x @ proj_w + b, LayerNorm over PD ----------------
// one block per (b,k) row; 64 threads = PD
__global__ void frontend_kernel(const float* __restrict__ x,
                                const float* __restrict__ proj_w,
                                const float* __restrict__ proj_b,
                                const float* __restrict__ ln_w,
                                const float* __restrict__ ln_b,
                                float* __restrict__ hln) {
  int row = blockIdx.x;            // b*KSEQ + k
  int j = threadIdx.x;             // 0..63
  __shared__ float xs[DIN];
  __shared__ float red[PDIM];
  if (j < DIN) xs[j] = x[(long)row * DIN + j];
  __syncthreads();
  float acc = proj_b[j];
#pragma unroll
  for (int i = 0; i < DIN; ++i) acc += xs[i] * proj_w[i * PDIM + j];
  red[j] = acc; __syncthreads();
  for (int s = PDIM / 2; s > 0; s >>= 1) { if (j < s) red[j] += red[j + s]; __syncthreads(); }
  float mu = red[0] / PDIM; __syncthreads();
  float d = acc - mu;
  red[j] = d * d; __syncthreads();
  for (int s = PDIM / 2; s > 0; s >>= 1) { if (j < s) red[j] += red[j + s]; __syncthreads(); }
  float var = red[0] / PDIM;
  hln[(long)row * PDIM + j] = d * rsqrtf(var + 1e-5f) * ln_w[j] + ln_b[j];
}

// ---------------- patch embed: e[m,t,:] = p @ patch_w + patch_b ----------------
// one block per (m,t); 128 threads = DM
__global__ void embed_kernel(const float* __restrict__ hln,
                             const float* __restrict__ patch_w,
                             const float* __restrict__ patch_b,
                             float* __restrict__ e) {
  long idx = blockIdx.x;           // m*NPATCH + t
  int m = (int)(idx / NPATCH), t = (int)(idx % NPATCH);
  int b = m / PDIM, pd = m % PDIM;
  int dm = threadIdx.x;
  __shared__ float s[PLEN];
  if (dm < PLEN) s[dm] = hln[((long)b * KSEQ + (t * PSTRIDE + dm)) * PDIM + pd];
  __syncthreads();
  float acc = patch_b[dm];
#pragma unroll
  for (int jj = 0; jj < PLEN; ++jj) acc += s[jj] * patch_w[jj * DMODEL + dm];
  e[idx * DMODEL + dm] = acc;
}

// ---------------- RMSNorm -> bf16 ----------------
__global__ void rms_kernel(const float* __restrict__ e, const float* __restrict__ w,
                           __hip_bfloat16* __restrict__ xn) {
  long r = blockIdx.x;
  int dm = threadIdx.x;
  __shared__ float red[DMODEL];
  float v = e[r * DMODEL + dm];
  red[dm] = v * v; __syncthreads();
  for (int s = DMODEL / 2; s > 0; s >>= 1) { if (dm < s) red[dm] += red[dm + s]; __syncthreads(); }
  float scale = rsqrtf(red[0] / DMODEL + 1e-5f);
  xn[r * DMODEL + dm] = __float2bfloat16(v * scale * w[dm]);
}

// ---------------- bf16 WMMA GEMM:  D(M,N) = A(M,K) * BT(N,K)^T  [+ Cres] ----------------
// block = 256 threads = 8 waves. Each wave owns a 16 x (16*NT) tile: one A fragment is
// reused across NT accumulators (NT WMMAs per K-step for 2+2*NT B128 loads).
// Grid sized so total wave-tiles == 8 * gridDim.x exactly (EXEC all 1s everywhere).
template <int NT>
__global__ void gemm_bf16_kernel(const __hip_bfloat16* __restrict__ A,
                                 const __hip_bfloat16* __restrict__ BT,
                                 float* __restrict__ D,
                                 const float* __restrict__ Cres,
                                 int Kd, int Nout) {
  const int lane = threadIdx.x & 31;
  const int wave = threadIdx.x >> 5;
  const int colGroups = Nout / (16 * NT);
  const long gtile = (long)blockIdx.x * 8 + wave;
  const int rowTile = (int)(gtile / colGroups);
  const int colGroup = (int)(gtile % colGroups);
  const int m0 = rowTile << 4;
  const int n0 = colGroup * (16 * NT);
  const int half = lane >> 4;     // 0: lanes 0-15, 1: lanes 16-31
  const int l16 = lane & 15;
  const __hip_bfloat16* arow = A + (long)(m0 + l16) * Kd;  // A row for this lane
  const __hip_bfloat16* brow = BT + (long)(n0 + l16) * Kd; // first B column group
  v8f c[NT];
  if (Cres) {
#pragma unroll
    for (int j = 0; j < NT; ++j)
#pragma unroll
      for (int v = 0; v < 8; ++v)
        c[j][v] = Cres[(long)(m0 + v + half * 8) * Nout + n0 + 16 * j + l16];
  } else {
#pragma unroll
    for (int j = 0; j < NT; ++j)
#pragma unroll
      for (int v = 0; v < 8; ++v) c[j][v] = 0.0f;
  }
  for (int kk = 0; kk < Kd; kk += 32) {
    __builtin_prefetch((const void*)(arow + kk + 64), 0, 0);
    // 16-bit A 16x32 layout: lane<16 -> K {kk..kk+7, kk+16..kk+23}; lane>=16 -> +8
    ABFrag a;
    const int ka = kk + half * 8;
    a.u[0] = *reinterpret_cast<const uv4*>(arow + ka);
    a.u[1] = *reinterpret_cast<const uv4*>(arow + ka + 16);
    // 16-bit B 32x16 layout: lane<16 -> K {kk..kk+7, kk+16..kk+23}; lane>=16 -> +8
    const int kb = kk + half * 8;
#pragma unroll
    for (int j = 0; j < NT; ++j) {
      ABFrag b;
      const __hip_bfloat16* bj = brow + (long)(16 * j) * Kd;
      b.u[0] = *reinterpret_cast<const uv4*>(bj + kb);
      b.u[1] = *reinterpret_cast<const uv4*>(bj + kb + 16);
      c[j] = __builtin_amdgcn_wmma_f32_16x16x32_bf16(false, a.v, false, b.v,
                                                     (short)0, c[j], false, false);
    }
  }
#pragma unroll
  for (int j = 0; j < NT; ++j)
#pragma unroll
    for (int v = 0; v < 8; ++v)
      D[(long)(m0 + v + half * 8) * Nout + n0 + 16 * j + l16] = c[j][v];
}

// ---------------- depthwise causal conv (DC=4) + SiLU ----------------
// block per sequence m, 256 threads = channel d; sequential over t
__global__ void conv_silu_kernel(const float* __restrict__ uz,
                                 const float* __restrict__ conv_w,
                                 const float* __restrict__ conv_b,
                                 float* __restrict__ u,
                                 __hip_bfloat16* __restrict__ ubf) {
  int m = blockIdx.x;
  int d = threadIdx.x;
  float w0 = conv_w[d * DCONV + 0], w1 = conv_w[d * DCONV + 1];
  float w2 = conv_w[d * DCONV + 2], w3 = conv_w[d * DCONV + 3];
  float bias = conv_b[d];
  float xm3 = 0.f, xm2 = 0.f, xm1 = 0.f;
  for (int t = 0; t < NPATCH; ++t) {
    long row = (long)m * NPATCH + t;
    float xt = uz[row * (2 * DINNER) + d];
    float uc = w0 * xm3 + w1 * xm2 + w2 * xm1 + w3 * xt + bias;
    xm3 = xm2; xm2 = xm1; xm1 = xt;
    float s = uc / (1.0f + __expf(-uc));   // SiLU
    u[row * DINNER + d] = s;
    ubf[row * DINNER + d] = __float2bfloat16(s);
  }
}

// ---------------- selective scan with fused delta = softplus(dt@dt_w+dt_b) and gate ----------------
// block per sequence m, 256 threads = channel d.
// Per step, the 40-float dbc row (dt[8] | B[16] | C[16]) is broadcast via LDS (double-buffered,
// one barrier per step); each thread computes its own delta from the staged dt values.
__global__ void scan_kernel(const float* __restrict__ u,
                            const float* __restrict__ dbc,
                            const float* __restrict__ uz,
                            const float* __restrict__ dt_w,
                            const float* __restrict__ dt_b,
                            const float* __restrict__ A_log,
                            const float* __restrict__ Dp,
                            __hip_bfloat16* __restrict__ ybf) {
  int m = blockIdx.x;
  int d = threadIdx.x;
  __shared__ float bc[2][XPAD];    // [0..7]=dt, [8..23]=B, [24..39]=C
  float A[DSTATE];
#pragma unroll
  for (int s = 0; s < DSTATE; ++s) A[s] = -__expf(A_log[d * DSTATE + s]);
  float dtw[DTRANK];
#pragma unroll
  for (int j = 0; j < DTRANK; ++j) dtw[j] = dt_w[j * DINNER + d];
  float dtb = dt_b[d];
  float Dpd = Dp[d];
  float h[DSTATE];
#pragma unroll
  for (int s = 0; s < DSTATE; ++s) h[s] = 0.0f;
  for (int t = 0; t < NPATCH; ++t) {
    long row = (long)m * NPATCH + t;
    int buf = t & 1;
    if (d < 2 * DSTATE + DTRANK) bc[buf][d] = dbc[row * XPAD + d];
    __syncthreads();
    // delta = softplus(dt @ dt_w + dt_b)
    float acc = dtb;
#pragma unroll
    for (int j = 0; j < DTRANK; ++j) acc += bc[buf][j] * dtw[j];
    float dt = (acc > 20.0f) ? acc : log1pf(__expf(acc));
    float ut = u[row * DINNER + d];
    float du = dt * ut;
    float y = 0.0f;
#pragma unroll
    for (int s = 0; s < DSTATE; ++s) {
      float dA = __expf(dt * A[s]);
      h[s] = dA * h[s] + du * bc[buf][DTRANK + s];
      y += h[s] * bc[buf][DTRANK + DSTATE + s];
    }
    float z = uz[row * (2 * DINNER) + DINNER + d];
    float g = z / (1.0f + __expf(-z));     // SiLU(z)
    ybf[row * DINNER + d] = __float2bfloat16((y + ut * Dpd) * g);
  }
}

// ---------------- final RMS + bb_w dot per sequence ----------------
__global__ void final_bb_kernel(const float* __restrict__ e,
                                const float* __restrict__ fnorm_w,
                                const float* __restrict__ bb_w,
                                const float* __restrict__ bb_b,
                                float* __restrict__ ymid) {
  int m = blockIdx.x;
  int dm = threadIdx.x;
  __shared__ float red[DMODEL];
  float fw = fnorm_w[dm];
  float acc = 0.0f;
  for (int t = 0; t < NPATCH; ++t) {
    long row = (long)m * NPATCH + t;
    float v = e[row * DMODEL + dm];
    red[dm] = v * v; __syncthreads();
    for (int s = DMODEL / 2; s > 0; s >>= 1) { if (dm < s) red[dm] += red[dm + s]; __syncthreads(); }
    float scale = rsqrtf(red[0] / DMODEL + 1e-5f);
    __syncthreads();
    acc += v * scale * fw * bb_w[t * DMODEL + dm];
  }
  red[dm] = acc; __syncthreads();
  for (int s = DMODEL / 2; s > 0; s >>= 1) { if (dm < s) red[dm] += red[dm + s]; __syncthreads(); }
  if (dm == 0) ymid[m] = red[0] + bb_b[0];
}

// ---------------- head: out[b,:] = ymid[b,:] @ head_w + head_b ----------------
__global__ void head_kernel(const float* __restrict__ ymid,
                            const float* __restrict__ head_w,
                            const float* __restrict__ head_b,
                            float* __restrict__ out) {
  int b = blockIdx.x;
  int pd = threadIdx.x;
  __shared__ float red[PDIM];
  float v = ymid[b * PDIM + pd];
  for (int c = 0; c < 2; ++c) {
    red[pd] = v * head_w[pd * 2 + c]; __syncthreads();
    for (int s = PDIM / 2; s > 0; s >>= 1) { if (pd < s) red[pd] += red[pd + s]; __syncthreads(); }
    if (pd == 0) out[b * 2 + c] = red[0] + head_b[c];
    __syncthreads();
  }
}

// ---------------- host orchestration ----------------
extern "C" void kernel_launch(void* const* d_in, const int* in_sizes, int n_in,
                              void* d_out, int out_size, void* d_ws, size_t ws_size,
                              hipStream_t stream) {
  const float* x       = (const float*)d_in[0];
  const float* proj_w  = (const float*)d_in[1];
  const float* proj_b  = (const float*)d_in[2];
  const float* ln_w    = (const float*)d_in[3];
  const float* ln_b    = (const float*)d_in[4];
  const float* patch_w = (const float*)d_in[5];
  const float* patch_b = (const float*)d_in[6];
  const float* in_w    = (const float*)d_in[7];
  const float* conv_w  = (const float*)d_in[8];
  const float* conv_b  = (const float*)d_in[9];
  const float* xproj_w = (const float*)d_in[10];
  const float* dt_w    = (const float*)d_in[11];
  const float* dt_b    = (const float*)d_in[12];
  const float* A_log   = (const float*)d_in[13];
  const float* Dp      = (const float*)d_in[14];
  const float* out_w   = (const float*)d_in[15];
  const float* norm_w  = (const float*)d_in[16];
  const float* fnorm_w = (const float*)d_in[17];
  const float* bb_w    = (const float*)d_in[18];
  const float* bb_b    = (const float*)d_in[19];
  const float* head_w  = (const float*)d_in[20];
  const float* head_b  = (const float*)d_in[21];
  float* out = (float*)d_out;

  char* ws = (char*)d_ws;
  size_t off = 0;
  auto take = [&](size_t bytes) -> void* {
    void* p = ws + off;
    off += (bytes + 255) & ~(size_t)255;
    return p;
  };
  const size_t R = (size_t)ROWS;
  float*           hln   = (float*)take((size_t)BATCH * KSEQ * PDIM * 4);
  float*           e     = (float*)take(R * DMODEL * 4);
  __hip_bfloat16*  xn    = (__hip_bfloat16*)take(R * DMODEL * 2);
  float*           uz    = (float*)take(R * 2 * DINNER * 4);
  float*           u     = (float*)take(R * DINNER * 4);
  __hip_bfloat16*  ubf   = (__hip_bfloat16*)take(R * DINNER * 2);
  float*           dbc   = (float*)take(R * XPAD * 4);
  __hip_bfloat16*  ybf   = (__hip_bfloat16*)take(R * DINNER * 2);
  __hip_bfloat16*  wInT  = (__hip_bfloat16*)take((size_t)NLAYER * 512 * 128 * 2);
  __hip_bfloat16*  wXpT  = (__hip_bfloat16*)take((size_t)NLAYER * XPAD * DINNER * 2);
  __hip_bfloat16*  wOutT = (__hip_bfloat16*)take((size_t)NLAYER * DMODEL * DINNER * 2);
  float*           ymid  = (float*)take((size_t)MSEQ * 4);

  // -------- weight prep (fp32 -> bf16, transposed, padded) --------
  for (int l = 0; l < NLAYER; ++l) {
    {
      long total = 512L * 128; int blocks = (int)((total + 255) / 256);
      transpose_bf16_kernel<<<blocks, 256, 0, stream>>>(
          in_w + (size_t)l * DMODEL * 2 * DINNER, wInT + (size_t)l * 512 * 128, 128, 512, 512);
    }
    {
      long total = (long)XPAD * DINNER; int blocks = (int)((total + 255) / 256);
      transpose_bf16_kernel<<<blocks, 256, 0, stream>>>(
          xproj_w + (size_t)l * DINNER * 40, wXpT + (size_t)l * XPAD * DINNER, DINNER, 40, XPAD);
    }
    {
      long total = (long)DMODEL * DINNER; int blocks = (int)((total + 255) / 256);
      transpose_bf16_kernel<<<blocks, 256, 0, stream>>>(
          out_w + (size_t)l * DINNER * DMODEL, wOutT + (size_t)l * DMODEL * DINNER, DINNER, DMODEL, DMODEL);
    }
  }

  // -------- frontend + patch embed --------
  frontend_kernel<<<BATCH * KSEQ, PDIM, 0, stream>>>(x, proj_w, proj_b, ln_w, ln_b, hln);
  embed_kernel<<<ROWS, DMODEL, 0, stream>>>(hln, patch_w, patch_b, e);

  // -------- mamba layers --------
  const int rowTiles = ROWS / 16; // 16256
  for (int l = 0; l < NLAYER; ++l) {
    rms_kernel<<<ROWS, DMODEL, 0, stream>>>(e, norm_w + (size_t)l * DMODEL, xn);

    // uz = xn @ in_w  (R x 512); 16x64 tiles: 512/64=8 groups -> 130048 wave-tiles
    {
      int gtiles = rowTiles * (512 / 64);
      gemm_bf16_kernel<4><<<gtiles / 8, 256, 0, stream>>>(
          xn, wInT + (size_t)l * 512 * 128, uz, nullptr, DMODEL, 512);
    }

    conv_silu_kernel<<<MSEQ, DINNER, 0, stream>>>(
        uz, conv_w + (size_t)l * DINNER * DCONV, conv_b + (size_t)l * DINNER, u, ubf);

    // dbc = u @ xproj_w  (R x 48, cols 40..47 zero); 16x48 tiles
    {
      int gtiles = rowTiles * (XPAD / 48);
      gemm_bf16_kernel<3><<<gtiles / 8, 256, 0, stream>>>(
          ubf, wXpT + (size_t)l * XPAD * DINNER, dbc, nullptr, DINNER, XPAD);
    }

    // scan with fused delta computation and SiLU gate
    scan_kernel<<<MSEQ, DINNER, 0, stream>>>(
        u, dbc, uz, dt_w + (size_t)l * DTRANK * DINNER, dt_b + (size_t)l * DINNER,
        A_log + (size_t)l * DINNER * DSTATE, Dp + (size_t)l * DINNER, ybf);

    // e += ybf @ out_w  (residual fused as WMMA C operand); 16x64 tiles: 128/64=2 groups
    {
      int gtiles = rowTiles * (DMODEL / 64);
      gemm_bf16_kernel<4><<<gtiles / 8, 256, 0, stream>>>(
          ybf, wOutT + (size_t)l * DMODEL * DINNER, e, e, DINNER, DMODEL);
    }
  }

  // -------- final norm + bottleneck + head --------
  final_bb_kernel<<<MSEQ, DMODEL, 0, stream>>>(e, fnorm_w, bb_w, bb_b, ymid);
  head_kernel<<<BATCH, PDIM, 0, stream>>>(ymid, head_w, head_b, out);

  (void)in_sizes; (void)n_in; (void)out_size; (void)ws_size;
}
